// DeformableEnhancedSpatialAttention_60189671686360
// MI455X (gfx1250) — matrix operand order repaired and checked by reading, not compiled, not run on hardware
//
#include <hip/hip_runtime.h>
#include <math.h>

// ---------------------------------------------------------------------------
// MI455X (gfx1250) implementation. wave32, WMMA 16x16x32 f16->f32.
// Memory-bound pipeline (~2.5 GB HBM floor -> ~110us @ 23.3 TB/s), so all
// intermediates that feed GEMMs are stored f16, and the whole tail
// (upsample + wb + add + w2 + sigmoid + x*c2) is fused into one kernel.
// ---------------------------------------------------------------------------

typedef __attribute__((ext_vector_type(16))) _Float16 v16h;
typedef __attribute__((ext_vector_type(8)))  float    v8f;

#define WMMA_F16(a, b, c) \
  __builtin_amdgcn_wmma_f32_16x16x32_f16(false, (a), false, (b), (short)0, (c), false, false)

// A-operand element->K map for 16x16x32 f16 (contiguous, matches B):
// lanes 0-15 hold K {0..15}; lanes 16-31 hold K {16..31}
__device__ __forceinline__ int a_k_of(int e, int hi) {
  return 16 * hi + e;
}

// ---------------------------------------------------------------------------
// Kernel 1: c0 = conv1x1(x, w1) + b1      (256 -> 64), store f16 NCHW
// GEMM: M = pixels, N = 64, K = 256. 8 waves/WG, 128-pixel tiles.
// ---------------------------------------------------------------------------
__global__ __launch_bounds__(256) void k_conv1x1_in(
    const float* __restrict__ x, const float* __restrict__ w1,
    const float* __restrict__ b1, _Float16* __restrict__ c0h) {
  constexpr int P = 224 * 224;
  constexpr int BS = 260;  // padded K-stride for B tile (bank-conflict pad)
  __shared__ _Float16 Bw[64 * BS];

  const int n = blockIdx.y, tile = blockIdx.x, t = threadIdx.x;
  for (int i = t; i < 64 * 256; i += 256) {
    int co = i >> 8, ci = i & 255;
    Bw[co * BS + ci] = (_Float16)w1[i];
  }
  __syncthreads();

  const int wv = t >> 5, l = t & 31, lm = l & 15, hi = l >> 4;
  const int pix0 = tile * 128 + wv * 16;
  const int apix = pix0 + lm;  // A-row pixel for this lane
  const float* xn = x + (size_t)n * 256 * P;

  v8f acc[4] = {};
  for (int kb = 0; kb < 256; kb += 32) {
    v16h a;
#pragma unroll
    for (int e = 0; e < 16; ++e) {
      int k = kb + a_k_of(e, hi);
      a[e] = (_Float16)xn[(size_t)k * P + apix];
    }
#pragma unroll
    for (int ns = 0; ns < 4; ++ns) {
      v16h b;
#pragma unroll
      for (int e = 0; e < 16; ++e) b[e] = Bw[(ns * 16 + lm) * BS + kb + 16 * hi + e];
      acc[ns] = WMMA_F16(a, b, acc[ns]);
    }
  }
  // D layout: lane holds cout = ns*16 + lm, pixels pix0 + 8*hi + r
  _Float16* dn = c0h + (size_t)n * 64 * P;
#pragma unroll
  for (int ns = 0; ns < 4; ++ns) {
    int co = ns * 16 + lm;
    float bias = b1[co];
#pragma unroll
    for (int r = 0; r < 8; ++r)
      dn[(size_t)co * P + pix0 + 8 * hi + r] = (_Float16)(acc[ns][r] + bias);
  }
}

// ---------------------------------------------------------------------------
// Kernel 2: t1 = conv3x3 stride2 valid (c0 f16 -> 111x111 f32) + bs
// GEMM: M = out pixels, N = 64, K = 576 (ci*9+tap). im2col staged in LDS.
// ---------------------------------------------------------------------------
__global__ __launch_bounds__(256) void k_conv3x3_s2(
    const _Float16* __restrict__ c0h, const float* __restrict__ wsp,
    const float* __restrict__ bs, float* __restrict__ t1) {
  constexpr int P = 224 * 224, W = 224, HO = 111, PO = 111 * 111, KD = 576;
  constexpr int BS = 580, AS = 36;
  __shared__ _Float16 Bw[64 * BS];   // 74,240 B
  __shared__ _Float16 At[128 * AS];  //  9,216 B

  const int n = blockIdx.y, tile = blockIdx.x, t = threadIdx.x;
  for (int i = t; i < 64 * KD; i += 256) {
    int o = i / KD, k = i - o * KD;
    Bw[o * BS + k] = (_Float16)wsp[i];
  }
  __syncthreads();

  const int wv = t >> 5, l = t & 31, lm = l & 15, hi = l >> 4;
  const int kk = t & 31, chunk = t >> 5;  // staging roles
  const _Float16* src = c0h + (size_t)n * 64 * P;

  v8f acc[4] = {};
  for (int kb = 0; kb < KD; kb += 32) {
    {  // stage A tile (im2col)
      int k = kb + kk;
      int ci = k / 9, tap = k - ci * 9, ky = tap / 3, kx = tap - (tap / 3) * 3;
      const _Float16* pl = src + (size_t)ci * P;
#pragma unroll
      for (int i = 0; i < 16; ++i) {
        int mloc = chunk * 16 + i;
        int g = tile * 128 + mloc;
        _Float16 v = (_Float16)0.f;
        if (g < PO) {
          int oy = g / HO, ox = g - oy * HO;
          v = pl[(2 * oy + ky) * W + (2 * ox + kx)];
        }
        At[mloc * AS + kk] = v;
      }
    }
    __syncthreads();
    v16h a;
#pragma unroll
    for (int e = 0; e < 16; ++e) a[e] = At[(wv * 16 + lm) * AS + a_k_of(e, hi)];
#pragma unroll
    for (int ns = 0; ns < 4; ++ns) {
      v16h b;
#pragma unroll
      for (int e = 0; e < 16; ++e) b[e] = Bw[(ns * 16 + lm) * BS + kb + 16 * hi + e];
      acc[ns] = WMMA_F16(a, b, acc[ns]);
    }
    __syncthreads();
  }
  float* dst = t1 + (size_t)n * 64 * PO;
#pragma unroll
  for (int ns = 0; ns < 4; ++ns) {
    int co = ns * 16 + lm;
    float bias = bs[co];
#pragma unroll
    for (int r = 0; r < 8; ++r) {
      int g = tile * 128 + wv * 16 + 8 * hi + r;
      if (g < PO) dst[(size_t)co * PO + g] = acc[ns][r] + bias;
    }
  }
}

// ---------------------------------------------------------------------------
// Kernel 3: 7x7 stride-3 max pool (111 -> 35), trivially memory bound.
// ---------------------------------------------------------------------------
__global__ __launch_bounds__(256) void k_maxpool(const float* __restrict__ t1,
                                                 float* __restrict__ p0) {
  constexpr int HO = 111, PO = 111 * 111, H3 = 35, P3 = 35 * 35;
  int idx = blockIdx.x * 256 + threadIdx.x;
  if (idx >= 16 * 64 * P3) return;
  int p = idx % P3, nc = idx / P3;
  int oy = p / H3, ox = p - oy * H3;
  const float* pl = t1 + (size_t)nc * PO;
  float mx = -INFINITY;
  for (int dy = 0; dy < 7; ++dy) {
    const float* row = pl + (3 * oy + dy) * HO + 3 * ox;
#pragma unroll
    for (int dx = 0; dx < 7; ++dx) mx = fmaxf(mx, row[dx]);
  }
  p0[idx] = mx;
}

// ---------------------------------------------------------------------------
// Kernel 4 (x3): deformable conv at 35x35: offsets conv (VALU, tiny N=18),
// bilinear-sampled im2col tile in LDS, einsum as WMMA (M=pix, N=64, K=576),
// bias + ReLU.
// ---------------------------------------------------------------------------
__global__ __launch_bounds__(256) void k_deform(
    const float* __restrict__ pin, const float* __restrict__ wo,
    const float* __restrict__ bo, const float* __restrict__ wd,
    const float* __restrict__ bd, float* __restrict__ pout) {
  constexpr int H3 = 35, P3 = 1225, KD = 576, BS = 580, AS = 36;
  __shared__ _Float16 Bw[64 * BS];          // phase1: aliased as f32 wo (41.5KB<74.2KB)
  __shared__ float offy[128 * 9], offx[128 * 9];
  __shared__ _Float16 At[128 * AS];

  const int n = blockIdx.y, tile = blockIdx.x, t = threadIdx.x;
  const float* in_n = pin + (size_t)n * 64 * P3;

  // ---- phase 1: offsets conv (18 = 9 taps * {y,x}) ----
  float* woL = (float*)Bw;
  for (int i = t; i < 18 * 576; i += 256) woL[i] = wo[i];
  __syncthreads();
  for (int task = t; task < 128 * 9; task += 256) {
    int m = task / 9, k = task - (task / 9) * 9;
    int g = tile * 128 + m;
    float ay = 0.f, ax = 0.f;
    if (g < P3) {
      int oy = g / H3, ox = g - (g / H3) * H3;
      const float* wy_ = woL + (2 * k) * 576;
      const float* wx_ = woL + (2 * k + 1) * 576;
      for (int ci = 0; ci < 64; ++ci) {
        const float* pl = in_n + ci * P3;
#pragma unroll
        for (int t2 = 0; t2 < 9; ++t2) {
          int ky2 = t2 / 3, kx2 = t2 - (t2 / 3) * 3;
          int iy = oy - 1 + ky2, ix = ox - 1 + kx2;
          float v = (iy >= 0 && iy < H3 && ix >= 0 && ix < H3) ? pl[iy * H3 + ix] : 0.f;
          ay += v * wy_[ci * 9 + t2];
          ax += v * wx_[ci * 9 + t2];
        }
      }
      int ky = k / 3, kx = k - (k / 3) * 3;
      ay += bo[2 * k] + (float)(oy - 1 + ky);
      ax += bo[2 * k + 1] + (float)(ox - 1 + kx);
    }
    offy[task] = ay;
    offx[task] = ax;
  }
  __syncthreads();

  // ---- phase 2: wd -> f16 LDS (overwrites wo region) ----
  for (int i = t; i < 64 * KD; i += 256) {
    int o = i / KD, k = i - (i / KD) * KD;
    Bw[o * BS + k] = (_Float16)wd[i];
  }
  __syncthreads();

  // ---- phase 3: sampled im2col + WMMA ----
  const int wv = t >> 5, l = t & 31, lm = l & 15, hi = l >> 4;
  const int kk = t & 31, chunk = t >> 5;
  v8f acc[4] = {};
  for (int kb = 0; kb < KD; kb += 32) {
    {
      int k = kb + kk;
      int ci = k / 9, tap = k - (k / 9) * 9;
      const float* pl = in_n + ci * P3;
#pragma unroll
      for (int i = 0; i < 16; ++i) {
        int mloc = chunk * 16 + i;
        int g = tile * 128 + mloc;
        float v = 0.f;
        if (g < P3) {
          float ys = offy[mloc * 9 + tap], xs = offx[mloc * 9 + tap];
          float y0f = floorf(ys), x0f = floorf(xs);
          int y0 = (int)y0f, x0 = (int)x0f;
          float wy = ys - y0f, wx = xs - x0f;
          bool yv0 = (y0 >= 0) && (y0 < H3), yv1 = (y0 + 1 >= 0) && (y0 + 1 < H3);
          bool xv0 = (x0 >= 0) && (x0 < H3), xv1 = (x0 + 1 >= 0) && (x0 + 1 < H3);
          float v00 = (yv0 && xv0) ? pl[y0 * H3 + x0] : 0.f;
          float v01 = (yv0 && xv1) ? pl[y0 * H3 + x0 + 1] : 0.f;
          float v10 = (yv1 && xv0) ? pl[(y0 + 1) * H3 + x0] : 0.f;
          float v11 = (yv1 && xv1) ? pl[(y0 + 1) * H3 + x0 + 1] : 0.f;
          v = v00 * (1.f - wy) * (1.f - wx) + v01 * (1.f - wy) * wx +
              v10 * wy * (1.f - wx) + v11 * wy * wx;
        }
        At[mloc * AS + kk] = (_Float16)v;
      }
    }
    __syncthreads();
    v16h a;
#pragma unroll
    for (int e = 0; e < 16; ++e) a[e] = At[(wv * 16 + lm) * AS + a_k_of(e, hi)];
#pragma unroll
    for (int ns = 0; ns < 4; ++ns) {
      v16h b;
#pragma unroll
      for (int e = 0; e < 16; ++e) b[e] = Bw[(ns * 16 + lm) * BS + kb + 16 * hi + e];
      acc[ns] = WMMA_F16(a, b, acc[ns]);
    }
    __syncthreads();
  }
  float* out_n = pout + (size_t)n * 64 * P3;
#pragma unroll
  for (int ns = 0; ns < 4; ++ns) {
    int co = ns * 16 + lm;
    float bias = bd[co];
#pragma unroll
    for (int r = 0; r < 8; ++r) {
      int g = tile * 128 + wv * 16 + 8 * hi + r;
      if (g < P3) out_n[(size_t)co * P3 + g] = fmaxf(acc[ns][r] + bias, 0.f);
    }
  }
}

// ---------------------------------------------------------------------------
// Kernel 5 (fused tail): per 128-pixel tile:
//   s = conv1x1(c0, wb)+bb + bilinear_up(d3)          (WMMA K=64, N=64)
//   c2 = sigmoid(conv1x1(s, w2)+b2)                   (WMMA K=64, N=256)
//   out = x * c2
// Avoids ~1.2 GB of HBM traffic vs. materializing s / c1_up / c2.
// ---------------------------------------------------------------------------
__global__ __launch_bounds__(256) void k_final(
    const float* __restrict__ x, const _Float16* __restrict__ c0h,
    const float* __restrict__ wb, const float* __restrict__ bb,
    const float* __restrict__ d3, const float* __restrict__ w2,
    const float* __restrict__ b2, float* __restrict__ out) {
  constexpr int P = 224 * 224, W = 224, H3 = 35, P3 = 1225;
  __shared__ _Float16 W2s[256 * 68];  // 34,816 B
  __shared__ _Float16 Wbs[64 * 68];   //  8,704 B
  __shared__ _Float16 St[128 * 68];   // 17,408 B (s tile, f16, K-contig rows)

  const int n = blockIdx.y, tile = blockIdx.x, t = threadIdx.x;

  // prefetch the x tile we will re-read in the epilogue (global_prefetch_b8)
  {
    const float* xb = x + (size_t)n * 256 * P + tile * 128;
    const char* pf = (const char*)(xb + (size_t)t * P);
    __builtin_prefetch(pf, 0, 0);
    __builtin_prefetch(pf + 256, 0, 0);
  }
  for (int i = t; i < 64 * 64; i += 256) {
    int o = i >> 6, k = i & 63;
    Wbs[o * 68 + k] = (_Float16)wb[i];
  }
  for (int i = t; i < 256 * 64; i += 256) {
    int o = i >> 6, k = i & 63;
    W2s[o * 68 + k] = (_Float16)w2[i];
  }
  __syncthreads();

  const int wv = t >> 5, l = t & 31, lm = l & 15, hi = l >> 4;
  const int pix0 = tile * 128 + wv * 16;
  const _Float16* cn = c0h + (size_t)n * 64 * P;

  // ---- branch GEMM: bacc = c0 x wb ----
  v8f bacc[4] = {};
  for (int kb = 0; kb < 64; kb += 32) {
    v16h a;
#pragma unroll
    for (int e = 0; e < 16; ++e) {
      int k = kb + a_k_of(e, hi);
      a[e] = cn[(size_t)k * P + pix0 + lm];
    }
#pragma unroll
    for (int ns = 0; ns < 4; ++ns) {
      v16h b;
#pragma unroll
      for (int e = 0; e < 16; ++e) b[e] = Wbs[(ns * 16 + lm) * 68 + kb + 16 * hi + e];
      bacc[ns] = WMMA_F16(a, b, bacc[ns]);
    }
  }
#pragma unroll
  for (int ns = 0; ns < 4; ++ns) {
    int co = ns * 16 + lm;
    float bias = bb[co];
#pragma unroll
    for (int r = 0; r < 8; ++r)
      St[(wv * 16 + 8 * hi + r) * 68 + co] = (_Float16)(bacc[ns][r] + bias);
  }
  __syncthreads();

  // ---- add bilinear upsample of d3 (35x35 -> 224x224), d3 is L2-resident ----
  const float* dn = d3 + (size_t)n * 64 * P3;
  for (int e = t; e < 64 * 128; e += 256) {
    int c = e >> 7, mloc = e & 127;
    int g = tile * 128 + mloc;
    int oy = g / W, ox = g - (g / W) * W;
    float sy = fmaxf((oy + 0.5f) * (35.f / 224.f) - 0.5f, 0.f);
    int y0 = min((int)sy, 34), y1 = min(y0 + 1, 34);
    float fy = sy - (float)y0;
    float sx = fmaxf((ox + 0.5f) * (35.f / 224.f) - 0.5f, 0.f);
    int x0 = min((int)sx, 34), x1 = min(x0 + 1, 34);
    float fx = sx - (float)x0;
    const float* pl = dn + c * P3;
    float up = pl[y0 * H3 + x0] * (1.f - fy) * (1.f - fx) +
               pl[y0 * H3 + x1] * (1.f - fy) * fx +
               pl[y1 * H3 + x0] * fy * (1.f - fx) + pl[y1 * H3 + x1] * fy * fx;
    St[mloc * 68 + c] = (_Float16)((float)St[mloc * 68 + c] + up);
  }
  __syncthreads();

  // ---- second GEMM: acc = s x w2  (N = 256) ----
  v8f acc[16] = {};
  for (int kb = 0; kb < 64; kb += 32) {
    v16h a;
#pragma unroll
    for (int e = 0; e < 16; ++e) a[e] = St[(wv * 16 + lm) * 68 + kb + a_k_of(e, hi)];
#pragma unroll
    for (int ns = 0; ns < 16; ++ns) {
      v16h b;
#pragma unroll
      for (int e = 0; e < 16; ++e) b[e] = W2s[(ns * 16 + lm) * 68 + kb + 16 * hi + e];
      acc[ns] = WMMA_F16(a, b, acc[ns]);
    }
  }

  // ---- sigmoid gate and multiply with x ----
  const float* xn = x + (size_t)n * 256 * P;
  float* on = out + (size_t)n * 256 * P;
#pragma unroll
  for (int ns = 0; ns < 16; ++ns) {
    int co = ns * 16 + lm;
    float bias = b2[co];
#pragma unroll
    for (int r = 0; r < 8; ++r) {
      int pix = pix0 + 8 * hi + r;
      float c2 = 1.f / (1.f + __expf(-(acc[ns][r] + bias)));
      on[(size_t)co * P + pix] = xn[(size_t)co * P + pix] * c2;
    }
  }
}

// ---------------------------------------------------------------------------
// Host-side launch
// ---------------------------------------------------------------------------
extern "C" void kernel_launch(void* const* d_in, const int* in_sizes, int n_in,
                              void* d_out, int out_size, void* d_ws, size_t ws_size,
                              hipStream_t stream) {
  const float* x   = (const float*)d_in[0];
  const float* w1  = (const float*)d_in[1];
  const float* b1  = (const float*)d_in[2];
  const float* wb  = (const float*)d_in[3];
  const float* bb  = (const float*)d_in[4];
  const float* wsc = (const float*)d_in[5];
  const float* bs  = (const float*)d_in[6];
  const float* wo1 = (const float*)d_in[7],  *bo1 = (const float*)d_in[8];
  const float* wd1 = (const float*)d_in[9],  *bd1 = (const float*)d_in[10];
  const float* wo2 = (const float*)d_in[11], *bo2 = (const float*)d_in[12];
  const float* wd2 = (const float*)d_in[13], *bd2 = (const float*)d_in[14];
  const float* wo3 = (const float*)d_in[15], *bo3 = (const float*)d_in[16];
  const float* wd3 = (const float*)d_in[17], *bd3 = (const float*)d_in[18];
  const float* w2  = (const float*)d_in[19], *b2 = (const float*)d_in[20];
  float* out = (float*)d_out;

  // workspace layout (~163.3 MB)
  char* wsp = (char*)d_ws;
  _Float16* c0h = (_Float16*)wsp;                                   // 102,760,448 B
  float* t1 = (float*)(wsp + 102760448);                            //  50,466,816 B
  float* pA = (float*)(wsp + 102760448 + 50466816);                 //   5,017,600 B
  float* pB = (float*)(wsp + 102760448 + 50466816 + 5017600);       //   5,017,600 B

  dim3 blk(256, 1, 1);
  k_conv1x1_in<<<dim3(392, 16, 1), blk, 0, stream>>>(x, w1, b1, c0h);
  k_conv3x3_s2<<<dim3(97, 16, 1), blk, 0, stream>>>(c0h, wsc, bs, t1);
  k_maxpool<<<dim3((16 * 64 * 1225 + 255) / 256, 1, 1), blk, 0, stream>>>(t1, pA);
  k_deform<<<dim3(10, 16, 1), blk, 0, stream>>>(pA, wo1, bo1, wd1, bd1, pB);
  k_deform<<<dim3(10, 16, 1), blk, 0, stream>>>(pB, wo2, bo2, wd2, bd2, pA);
  k_deform<<<dim3(10, 16, 1), blk, 0, stream>>>(pA, wo3, bo3, wd3, bd3, pB);
  k_final<<<dim3(392, 16, 1), blk, 0, stream>>>(x, c0h, wb, bb, pB, w2, b2, out);
}
